// RNN_81088982548788
// MI455X (gfx1250) — matrix operand order repaired and checked by reading, not compiled, not run on hardware
//
#include <hip/hip_runtime.h>

// RNN: x_proj GEMM -> persistent-scan recurrence -> output GEMM.
// Matmuls via V_WMMA_F32_16X16X32_BF16 (bf16 in, f32 accumulate).
// GEMM staging uses double-buffered LDS + GLOBAL_LOAD_ASYNC_TO_LDS_B128.
//
// d_ws layout:
//   [0,   2MB)            Wt_i2h  bf16, transposed  [n][k]
//   [2MB, 4MB)            Wt_h2h  bf16, transposed  [n][k]
//   [4MB, 6MB)            Wt_out  bf16, transposed  [n][k]
//   [6MB, 6MB+4KB)        sync counters (grid barrier)
//   [6MB+4KB, +67.25MB)   h_hist bf16 [513][64][1024]  (slot 0 = h_{-1} = 0)
//
// d_out layout: [0, 128MB) hidden_states f32 (first used as x_proj scratch,
// overwritten in place by relu(h)); [128MB, 256MB) output_states f32.

#define HD     1024
#define BATCH  64
#define SEQT   512
#define MROWS  (BATCH * SEQT)   // 32768

typedef __attribute__((ext_vector_type(4)))  unsigned int u32x4;
typedef __attribute__((ext_vector_type(4)))  float        f32x4;
typedef __attribute__((ext_vector_type(8)))  float        f32x8;
typedef __attribute__((ext_vector_type(16))) __bf16       bf16x16;

union FragU { bf16x16 v; u32x4 q[2]; };

__device__ __forceinline__ unsigned short f32_to_bf16(float f) {
  unsigned int u = __builtin_bit_cast(unsigned int, f);
  unsigned int r = 0x7FFFu + ((u >> 16) & 1u);   // round-to-nearest-even
  return (unsigned short)((u + r) >> 16);
}

__device__ __forceinline__ unsigned int pack2(float lo, float hi) {
  return (unsigned int)f32_to_bf16(lo) | ((unsigned int)f32_to_bf16(hi) << 16);
}

__device__ __forceinline__ f32x8 wmma_bf16(FragU a, FragU b, f32x8 c) {
  return __builtin_amdgcn_wmma_f32_16x16x32_bf16(
      false, a.v, false, b.v, (short)0, c, false, false);
}

// CDNA5 async copy: 16B global -> LDS per lane, tracked with ASYNCcnt.
__device__ __forceinline__ void async_copy_b128(const void* gptr, void* lptr) {
  unsigned int lds = (unsigned int)(size_t)lptr;  // low 32 bits = LDS offset
  asm volatile("global_load_async_to_lds_b128 %0, %1, off"
               :
               : "v"(lds), "v"(gptr)
               : "memory");
}

__device__ __forceinline__ void wait_async0() {
  asm volatile("s_wait_asynccnt 0x0" ::: "memory");
}

// Sense-reversing grid barrier over nwg persistent workgroups.
// sync[0] = arrival counter, sync[1] = generation.
__device__ __forceinline__ void grid_barrier(unsigned int* sync, unsigned int nwg) {
  __syncthreads();
  if (threadIdx.x == 0) {
    unsigned int gen  = __atomic_load_n(&sync[1], __ATOMIC_ACQUIRE);
    unsigned int prev = __atomic_fetch_add(&sync[0], 1u, __ATOMIC_ACQ_REL);
    if (prev + 1u == nwg) {
      __atomic_store_n(&sync[0], 0u, __ATOMIC_RELAXED);
      __atomic_store_n(&sync[1], gen + 1u, __ATOMIC_RELEASE);
    } else {
      while (__atomic_load_n(&sync[1], __ATOMIC_ACQUIRE) == gen)
        __builtin_amdgcn_s_sleep(2);
    }
  }
  __syncthreads();
  // Every wave invalidates its L0/WGP$ before re-reading cross-WGP data.
  __builtin_amdgcn_fence(__ATOMIC_ACQUIRE, "agent");
}

// ---------------------------------------------------------------------------
// Kernel 0: transpose+convert the three 1024x1024 f32 weights into bf16
// column-major Wt[n][k]; also zero the grid-barrier state.
// ---------------------------------------------------------------------------
__global__ __launch_bounds__(256) void convert_weights(
    const float* __restrict__ Wi, const float* __restrict__ Wh,
    const float* __restrict__ Wo,
    unsigned short* __restrict__ WtI, unsigned short* __restrict__ WtH,
    unsigned short* __restrict__ WtO, unsigned int* __restrict__ sync) {
  size_t idx = (size_t)blockIdx.x * 256 + threadIdx.x;   // 0 .. 3*2^20-1
  unsigned sel = (unsigned)(idx >> 20);
  unsigned rem = (unsigned)(idx & 0xFFFFFu);
  unsigned n = rem >> 10, k = rem & 1023u;
  const float* src = (sel == 0) ? Wi : (sel == 1) ? Wh : Wo;
  unsigned short* dst = (sel == 0) ? WtI : (sel == 1) ? WtH : WtO;
  dst[rem] = f32_to_bf16(src[((size_t)k << 10) | n]);    // Wt[n][k] = W[k][n]
  if (blockIdx.x == 0 && threadIdx.x < 8) sync[threadIdx.x] = 0u;
}

// ---------------------------------------------------------------------------
// Tiled GEMM: out[M x 1024] = A[M x 1024] (x) Bt + bias1 (+ bias2)
//   A_F32:      A is f32 (converted to bf16 while staging, VGPR path),
//               else A is bf16 (async global->LDS copy).
//   TMAJOR_OUT: A rows are (t*64+b); remap output address to [b][t][n].
// Block tile 128x128, 8 waves (4 M x 2 N), wave tile 32x64, K step 32,
// double-buffered LDS staging.
// ---------------------------------------------------------------------------
template <bool A_F32>
__device__ __forceinline__ void stage_tile(
    const void* __restrict__ Ap, const unsigned short* __restrict__ Bt,
    int m0, int n0, int k0, int arow, int koff,
    unsigned short* AshBuf, unsigned short* BshBuf) {
#pragma unroll
  for (int s = 0; s < 2; ++s) {
    const int r = arow + s * 64;
    if (A_F32) {
      const float* ap = (const float*)Ap + (size_t)(m0 + r) * HD + k0 + koff;
      f32x4 f0 = *(const f32x4*)ap;
      f32x4 f1 = *(const f32x4*)(ap + 4);
      u32x4 q;
      q.x = pack2(f0.x, f0.y);
      q.y = pack2(f0.z, f0.w);
      q.z = pack2(f1.x, f1.y);
      q.w = pack2(f1.z, f1.w);
      *(u32x4*)&AshBuf[r * 40 + koff] = q;
    } else {
      async_copy_b128(
          (const unsigned short*)Ap + (size_t)(m0 + r) * HD + k0 + koff,
          &AshBuf[r * 40 + koff]);
    }
    async_copy_b128(Bt + (size_t)(n0 + r) * HD + k0 + koff,
                    &BshBuf[r * 40 + koff]);
  }
}

template <bool A_F32, bool TMAJOR_OUT>
__global__ __launch_bounds__(256) void gemm_rnn(
    const void* __restrict__ Ap, const unsigned short* __restrict__ Bt,
    const float* __restrict__ bias1, const float* __restrict__ bias2,
    float* __restrict__ out) {
  __shared__ unsigned short Ash[2][128 * 40];  // row-major   [m][k], pad 32->40
  __shared__ unsigned short Bsh[2][128 * 40];  // column-major[n][k], pad 32->40

  const int tid = threadIdx.x;
  const int n0 = blockIdx.x * 128;
  const int m0 = blockIdx.y * 128;
  const int lane = tid & 31, wave = tid >> 5;
  const int wm = wave >> 1, wn = wave & 1;        // 4 x 2 wave grid
  const int lm = lane & 15, lh = lane >> 4;

  const f32x8 vzero = {0.f, 0.f, 0.f, 0.f, 0.f, 0.f, 0.f, 0.f};
  f32x8 acc[2][4];
#pragma unroll
  for (int i = 0; i < 2; ++i)
#pragma unroll
    for (int j = 0; j < 4; ++j) acc[i][j] = vzero;

  const int arow = tid >> 2;            // 0..63
  const int koff = (tid & 3) * 8;       // 0,8,16,24

  stage_tile<A_F32>(Ap, Bt, m0, n0, 0, arow, koff, Ash[0], Bsh[0]);

  for (int kt = 0; kt < 32; ++kt) {
    wait_async0();        // our async writes into buffer (kt&1) have landed
    __syncthreads();      // everyone's writes visible; prev compute done
    const int cur = kt & 1;
    if (kt < 31)
      stage_tile<A_F32>(Ap, Bt, m0, n0, (kt + 1) * 32, arow, koff,
                        Ash[cur ^ 1], Bsh[cur ^ 1]);

    FragU a[2], b[4];
#pragma unroll
    for (int mi = 0; mi < 2; ++mi) {
      const int m = wm * 32 + mi * 16 + lm;
      a[mi].q[0] = *(const u32x4*)&Ash[cur][m * 40 + lh * 8];
      a[mi].q[1] = *(const u32x4*)&Ash[cur][m * 40 + lh * 8 + 16];
    }
#pragma unroll
    for (int ni = 0; ni < 4; ++ni) {
      const int n = wn * 64 + ni * 16 + lm;
      b[ni].q[0] = *(const u32x4*)&Bsh[cur][n * 40 + lh * 8];
      b[ni].q[1] = *(const u32x4*)&Bsh[cur][n * 40 + lh * 8 + 16];
    }
#pragma unroll
    for (int mi = 0; mi < 2; ++mi)
#pragma unroll
      for (int ni = 0; ni < 4; ++ni)
        acc[mi][ni] = wmma_bf16(a[mi], b[ni], acc[mi][ni]);
  }

#pragma unroll
  for (int ni = 0; ni < 4; ++ni) {
    const int gn = n0 + wn * 64 + ni * 16 + lm;
    float bb = bias1[gn];
    if (bias2) bb += bias2[gn];
#pragma unroll
    for (int mi = 0; mi < 2; ++mi) {
#pragma unroll
      for (int r = 0; r < 8; ++r) {
        const int gm = m0 + wm * 32 + mi * 16 + lh * 8 + r;
        const float v = acc[mi][ni][r] + bb;
        size_t oa;
        if (TMAJOR_OUT) {
          const int tt = gm >> 6, bi = gm & 63;
          oa = (size_t)bi * (SEQT * HD) + (size_t)tt * HD + gn;
        } else {
          oa = (size_t)gm * HD + gn;
        }
        out[oa] = v;
      }
    }
  }
}

// ---------------------------------------------------------------------------
// Persistent recurrence: 16 WGs, WG g owns output columns [64g, 64g+64).
// Its 64x1024 bf16 slice of W_h2h stays in LDS (132 KB) for all 512 steps.
// h passes between steps through bf16 h_hist (L2 resident); steps separated
// by an atomic grid barrier. xp (f32, in d_out) is overwritten with relu(h).
// ---------------------------------------------------------------------------
__global__ __launch_bounds__(256) void rnn_scan(
    const unsigned short* __restrict__ WtH, unsigned short* __restrict__ h_hist,
    float* __restrict__ xp, unsigned int* __restrict__ sync) {
  extern __shared__ unsigned short Wsh[];   // [64][1032] bf16 (pad 1024->1032)
  const int tid = threadIdx.x;
  const int wg = blockIdx.x;
  const int n0 = wg * 64;

  // Preload weight slice: 64 cols x 1024 k, 8 bf16 per thread-slice.
#pragma unroll
  for (int s = 0; s < 32; ++s) {
    const int lin = s * 256 + tid;        // 0..8191
    const int n = lin >> 7;               // 0..63
    const int ko = (lin & 127) * 8;       // 0..1016
    *(u32x4*)&Wsh[n * 1032 + ko] =
        *(const u32x4*)&WtH[(size_t)(n0 + n) * HD + ko];
  }
  // Zero h_{-1} (slot 0): 32768 uints split across 16 WGs.
  {
    unsigned int* z = (unsigned int*)h_hist + (size_t)wg * 2048;
#pragma unroll
    for (int i = 0; i < 8; ++i) z[i * 256 + tid] = 0u;
  }
  __threadfence();
  grid_barrier(sync, gridDim.x);

  const int lane = tid & 31, wave = tid >> 5;
  const int wm = wave >> 1, wn = wave & 1;   // 4 M-waves x 2 N-waves
  const int lm = lane & 15, lh = lane >> 4;
  const int m = wm * 16 + lm;                // batch row for A fragment
  const f32x8 vzero = {0.f, 0.f, 0.f, 0.f, 0.f, 0.f, 0.f, 0.f};

  for (int t = 0; t < SEQT; ++t) {
    const unsigned short* hrow =
        h_hist + (size_t)t * (BATCH * HD) + (size_t)m * HD;
    f32x8 acc[2];
    acc[0] = vzero;
    acc[1] = vzero;
#pragma unroll 4
    for (int kt = 0; kt < 32; ++kt) {
      const int k0 = kt * 32;
      FragU a, b0, b1;
      a.q[0] = *(const u32x4*)(hrow + k0 + lh * 8);
      a.q[1] = *(const u32x4*)(hrow + k0 + lh * 8 + 16);
      const int kb = k0 + lh * 8;
      const int nb = (wn * 32 + lm) * 1032;
      b0.q[0] = *(const u32x4*)&Wsh[nb + kb];
      b0.q[1] = *(const u32x4*)&Wsh[nb + kb + 16];
      b1.q[0] = *(const u32x4*)&Wsh[nb + 16 * 1032 + kb];
      b1.q[1] = *(const u32x4*)&Wsh[nb + 16 * 1032 + kb + 16];
      acc[0] = wmma_bf16(a, b0, acc[0]);
      acc[1] = wmma_bf16(a, b1, acc[1]);
    }
    unsigned short* hnext = h_hist + (size_t)(t + 1) * (BATCH * HD);
#pragma unroll
    for (int ni = 0; ni < 2; ++ni) {
      const int gn = n0 + wn * 32 + ni * 16 + lm;
#pragma unroll
      for (int r = 0; r < 8; ++r) {
        const int bm = wm * 16 + lh * 8 + r;   // batch index 0..63
        const size_t xa = (size_t)bm * (SEQT * HD) + (size_t)t * HD + gn;
        float v = acc[ni][r] + xp[xa];
        v = fmaxf(v, 0.f);
        xp[xa] = v;                                  // hidden_states (f32)
        hnext[(size_t)bm * HD + gn] = f32_to_bf16(v);  // feeds next step + GEMM
      }
    }
    __threadfence();
    grid_barrier(sync, gridDim.x);
  }
}

// ---------------------------------------------------------------------------
extern "C" void kernel_launch(void* const* d_in, const int* in_sizes, int n_in,
                              void* d_out, int out_size, void* d_ws,
                              size_t ws_size, hipStream_t stream) {
  const float* seq   = (const float*)d_in[0];
  const float* W_i2h = (const float*)d_in[1];
  const float* b_i2h = (const float*)d_in[2];
  const float* W_h2h = (const float*)d_in[3];
  const float* b_h2h = (const float*)d_in[4];
  const float* W_out = (const float*)d_in[5];
  const float* b_out = (const float*)d_in[6];

  char* ws = (char*)d_ws;
  unsigned short* WtI = (unsigned short*)(ws);
  unsigned short* WtH = (unsigned short*)(ws + (2ull << 20));
  unsigned short* WtO = (unsigned short*)(ws + (4ull << 20));
  unsigned int* sync  = (unsigned int*)(ws + (6ull << 20));
  unsigned short* h_hist = (unsigned short*)(ws + (6ull << 20) + 4096);

  float* xp   = (float*)d_out;                         // x_proj -> hidden (f32)
  float* out2 = (float*)d_out + (size_t)MROWS * HD;    // output_states

  // Allow >64KB dynamic LDS for the persistent scan kernel (non-stream call,
  // safe under graph capture).
  (void)hipFuncSetAttribute((const void*)rnn_scan,
                            hipFuncAttributeMaxDynamicSharedMemorySize, 132096);

  // 0) weights f32 -> bf16 transposed; zero barrier state
  convert_weights<<<12288, 256, 0, stream>>>(W_i2h, W_h2h, W_out, WtI, WtH,
                                             WtO, sync);
  // 1) x_proj = seq @ W_i2h + (b_i2h + b_h2h)   [f32 into d_out hidden region]
  gemm_rnn<true, false><<<dim3(8, 256), 256, 0, stream>>>(
      (const void*)seq, WtI, b_i2h, b_h2h, xp);
  // 2) persistent scan: h_t = relu(xp_t + h_{t-1} @ W_h2h)
  rnn_scan<<<16, 256, 132096, stream>>>(WtH, h_hist, xp, sync);
  // 3) output_states = hidden @ W_out + b_out   (A = bf16 h_hist, t-major)
  gemm_rnn<false, true><<<dim3(8, 256), 256, 0, stream>>>(
      (const void*)(h_hist + (size_t)BATCH * HD), WtO, b_out, nullptr, out2);
}
